// ScaledDotProductAttention_16123307229484
// MI455X (gfx1250) — matrix operand order repaired and checked by reading, not compiled, not run on hardware
//
#include <hip/hip_runtime.h>
#include <hip/hip_bf16.h>

// ---------------------------------------------------------------------------
// Scaled dot-product attention for MI455X (gfx1250, wave32, WMMA).
// - All GEMMs on v_wmma_f32_16x16x32_bf16 (fp32 accumulate).
// - bf16 A operands: global_load_async_to_lds_b128 (ASYNCcnt).
// - bf16 B operands: Tensor Data Mover tensor_load_to_lds (TENSORcnt),
//   one descriptor per 128x32 tile, LDS padding done by the TDM.
// - fp32 operands converted in-register while staging (loads batched).
// Double-buffered LDS, one barrier per K step.
// ---------------------------------------------------------------------------

typedef __bf16 bf16;
typedef __attribute__((ext_vector_type(16))) __bf16 v16bf;
typedef __attribute__((ext_vector_type(8)))  __bf16 v8bf;
typedef __attribute__((ext_vector_type(4)))  __bf16 v4bf;
typedef __attribute__((ext_vector_type(8)))  float  v8f;
typedef __attribute__((ext_vector_type(4)))  float  v4f;
typedef __attribute__((ext_vector_type(4)))  int    v4i;
typedef __attribute__((ext_vector_type(8)))  int    v8i;

#define DIMN   1024
#define BATCH  4
#define SEQ    2048

#define BM 128
#define BN 128
#define BK 32
#define LDT 40                      // padded LDS row pitch (bf16 elems, 80 B)
#define BUFB (BM * LDT * 2)         // bytes per LDS buffer (10240)

// epilogue modes
#define EPI_BF16_OUT   0   // bf16 [m][n] + bias            (Q, K projections)
#define EPI_BF16_T_OUT 1   // bf16 [b][n][s] + bias         (V projection, transposed)
#define EPI_SCORE_OUT  2   // fp32 [m][n], *scale, mask->-inf
#define EPI_F32_OUT    3   // fp32 [m][n]                   (context)

__device__ __forceinline__ void wait_asynccnt0() {
#if __has_builtin(__builtin_amdgcn_s_wait_asynccnt)
  __builtin_amdgcn_s_wait_asynccnt(0);
#else
  asm volatile("s_wait_asynccnt 0" ::: "memory");
#endif
}

__device__ __forceinline__ void wait_tensorcnt0() {
#if __has_builtin(__builtin_amdgcn_s_wait_tensorcnt)
  __builtin_amdgcn_s_wait_tensorcnt(0);
#else
  asm volatile("s_wait_tensorcnt 0" ::: "memory");
#endif
}

// ---- sync stage: 128x32 fp32 tile -> bf16 LDS (loads batched, then cvt) ----
__device__ __forceinline__ void stage_f32(const float* __restrict__ g, int ld,
                                          bf16* __restrict__ s, int tid) {
  v4f t[4];
#pragma unroll
  for (int i = 0; i < 4; ++i) {
    int c = tid + i * 256, row = c >> 3, k4 = (c & 7) * 4;
    t[i] = *(const v4f*)(g + (size_t)row * ld + k4);
  }
#pragma unroll
  for (int i = 0; i < 4; ++i) {
    int c = tid + i * 256, row = c >> 3, k4 = (c & 7) * 4;
    *(v4bf*)(s + row * LDT + k4) = __builtin_convertvector(t[i], v4bf);
  }
}

// ---- async stage: 128x32 bf16 tile -> LDS via global_load_async_to_lds ----
__device__ __forceinline__ void stage_bf16_async(const bf16* __restrict__ g,
                                                 int ld, unsigned lds_base,
                                                 int tid) {
#pragma unroll
  for (int i = 0; i < 2; ++i) {
    int c = tid + i * 256, row = c >> 2, k8 = (c & 3) * 8;  // 16-B chunks
    const void* ga = (const void*)(g + (size_t)row * ld + k8);
    unsigned la = lds_base + (unsigned)(row * LDT + k8) * 2u;
    asm volatile("global_load_async_to_lds_b128 %0, %1, off"
                 :: "v"(la), "v"(ga)
                 : "memory");
  }
}

// ---- TDM stage: one descriptor moves the whole 128x32 bf16 tile -----------
// D# group0: count=1 | lds_addr | global_addr[56:0] | type=2
// D# group1: data_size=2B, pad_enable, pad_interval=16 DW (one 64-B row),
//            pad_amount=4 DW (16 B)  ->  reproduces the LDT=40 padded pitch.
//            tensor_dim0=BK, tensor_dim1=BM, tile_dim0=BK, tile_dim1=BM,
//            tensor_dim0_stride=ld (elements).
__device__ __forceinline__ void tdm_stage(const bf16* __restrict__ g,
                                          int ld_elems, unsigned lds_off) {
  unsigned long long ga = (unsigned long long)(uintptr_t)g;
  v4i g0 = {1, (int)lds_off, (int)(unsigned)(ga & 0xffffffffull),
            (int)(((ga >> 32) & 0x1ffffffull) | (2u << 30))};
  v8i g1 = {(1 << 16) | (1 << 20) | (3 << 22) | (3 << 25),  // dsz/pad cfg
            (BK << 16),        // atomic_addr=0 | tensor_dim0[15:0]
            (BM << 16),        // tensor_dim0[31:16]=0 | tensor_dim1[15:0]
            (BK << 16),        // tensor_dim1[31:16]=0 | tile_dim0
            BM,                // tile_dim1 | tile_dim2=0
            ld_elems,          // tensor_dim0_stride[31:0]
            0, 0};             // stride hi / tensor_dim1_stride = 0
  asm volatile("tensor_load_to_lds %0, %1" :: "s"(g0), "s"(g1) : "memory");
}

// ---------------------------------------------------------------------------
// Generic NT GEMM: C[m][n] = sum_k A[m][k] * B[n][k]  (both K-contiguous)
// ---------------------------------------------------------------------------
template <bool A_F32, bool B_F32, int EPI>
__global__ __launch_bounds__(256) void gemm_wmma(
    const void* __restrict__ Ap, const void* __restrict__ Bp,
    void* __restrict__ Outp, int lda, int ldb, int ldo, int K,
    long strideA, long strideB, long strideO,
    const float* __restrict__ bias,
    const int* __restrict__ mask, int strideMask, float scale) {
  __shared__ bf16 As[2][BM * LDT];
  __shared__ bf16 Bs[2][BN * LDT];

  const int tid  = threadIdx.x;
  const int lane = tid & 31;
  const int wave = tid >> 5;          // 8 waves
  const int wm   = wave & 3;          // 4 wave rows  (32 rows each)
  const int wn   = wave >> 2;         // 2 wave cols  (64 cols each)
  const int b    = blockIdx.z;
  const int m0   = blockIdx.y * BM;
  const int n0   = blockIdx.x * BN;

  const float* Af = (const float*)Ap;
  const bf16*  Ah = (const bf16*)Ap;
  const float* Bf = (const float*)Bp;
  const bf16*  Bh = (const bf16*)Bp;
  const size_t offA = (size_t)b * strideA + (size_t)m0 * lda;
  const size_t offB = (size_t)b * strideB + (size_t)n0 * ldb;

  const unsigned asbase = (unsigned)(uintptr_t)(&As[0][0]);
  const unsigned bsbase = (unsigned)(uintptr_t)(&Bs[0][0]);
  constexpr bool ASYNC_A = !A_F32;   // bf16 A -> per-lane async copy
  constexpr bool TDM_B   = !B_F32;   // bf16 B -> Tensor Data Mover

  v8f acc[2][4];
#pragma unroll
  for (int mt = 0; mt < 2; ++mt)
#pragma unroll
    for (int nt = 0; nt < 4; ++nt) acc[mt][nt] = (v8f)(0.f);

  const int arow = wm * 32 + (lane & 15);
  const int ka   = (lane >> 4) * 8;   // A lane K-chunk start (0 or 8)
  const int bcol = wn * 64 + (lane & 15);
  const int kb   = (lane >> 4) * 8;   // B lane K-chunk start (0 or 8)

  // prologue: fill buffer 0 for the first K tile
  if (ASYNC_A) stage_bf16_async(Ah + offA, lda, asbase, tid);
  if (TDM_B && wave == 0) tdm_stage(Bh + offB, ldb, bsbase);

  const int niter = K / BK;
  for (int i = 0; i < niter; ++i) {
    const int k0  = i * BK;
    const int cur = i & 1;
    bf16* Asc = &As[cur][0];
    bf16* Bsc = &Bs[cur][0];

    if (A_F32) stage_f32(Af + offA + k0, lda, Asc, tid);
    if (B_F32) stage_f32(Bf + offB + k0, ldb, Bsc, tid);
    if (i + 1 < niter) {  // global_prefetch_b8 for next fp32 tiles
      if (A_F32) __builtin_prefetch(Af + offA + k0 + BK + (size_t)(tid >> 1) * lda, 0, 1);
      if (B_F32) __builtin_prefetch(Bf + offB + k0 + BK + (size_t)(tid >> 1) * ldb, 0, 1);
    }
    if (ASYNC_A) wait_asynccnt0();
    if (TDM_B)   wait_tensorcnt0();
    __syncthreads();

    // refill the other buffer while this one is consumed
    if (i + 1 < niter) {
      if (ASYNC_A)
        stage_bf16_async(Ah + offA + k0 + BK, lda, asbase + (1 - cur) * BUFB, tid);
      if (TDM_B && wave == 0)
        tdm_stage(Bh + offB + k0 + BK, ldb, bsbase + (1 - cur) * BUFB);
    }

    v16bf afrag[2], bfrag[4];
#pragma unroll
    for (int mt = 0; mt < 2; ++mt) {
      v8bf lo = *(const v8bf*)(Asc + (arow + mt * 16) * LDT + ka);
      v8bf hi = *(const v8bf*)(Asc + (arow + mt * 16) * LDT + ka + 16);
      afrag[mt] = __builtin_shufflevector(lo, hi, 0, 1, 2, 3, 4, 5, 6, 7, 8, 9,
                                          10, 11, 12, 13, 14, 15);
    }
#pragma unroll
    for (int nt = 0; nt < 4; ++nt) {
      v8bf lo = *(const v8bf*)(Bsc + (bcol + nt * 16) * LDT + kb);
      v8bf hi = *(const v8bf*)(Bsc + (bcol + nt * 16) * LDT + kb + 16);
      bfrag[nt] = __builtin_shufflevector(lo, hi, 0, 1, 2, 3, 4, 5, 6, 7, 8, 9,
                                          10, 11, 12, 13, 14, 15);
    }
#pragma unroll
    for (int mt = 0; mt < 2; ++mt)
#pragma unroll
      for (int nt = 0; nt < 4; ++nt)
        acc[mt][nt] = __builtin_amdgcn_wmma_f32_16x16x32_bf16(
            false, afrag[mt], false, bfrag[nt], (short)0, acc[mt][nt], false,
            false);
  }

  // ---- epilogue ----
#pragma unroll
  for (int mt = 0; mt < 2; ++mt) {
#pragma unroll
    for (int nt = 0; nt < 4; ++nt) {
      const int n     = n0 + wn * 64 + nt * 16 + (lane & 15);
      const int mrow0 = m0 + wm * 32 + mt * 16 + (lane >> 4) * 8;
      if (EPI == EPI_BF16_OUT) {
        const float bv = bias[n];
        bf16* o = (bf16*)Outp;
#pragma unroll
        for (int r = 0; r < 8; ++r)
          o[(size_t)(mrow0 + r) * ldo + n] = (bf16)(acc[mt][nt][r] + bv);
      } else if (EPI == EPI_BF16_T_OUT) {
        const float bv = bias[n];
        const int bb = mrow0 / SEQ, s = mrow0 % SEQ;
        v8bf pk;
#pragma unroll
        for (int r = 0; r < 8; ++r) pk[r] = (bf16)(acc[mt][nt][r] + bv);
        *(v8bf*)((bf16*)Outp + (size_t)bb * DIMN * SEQ + (size_t)n * SEQ + s) = pk;
      } else if (EPI == EPI_SCORE_OUT) {
        float* o = (float*)Outp + (size_t)b * strideO;
        const bool mk = mask[b * strideMask + n] != 0;
#pragma unroll
        for (int r = 0; r < 8; ++r)
          o[(size_t)(mrow0 + r) * ldo + n] =
              mk ? -__builtin_inff() : acc[mt][nt][r] * scale;
      } else {
        float* o = (float*)Outp + (size_t)b * strideO;
#pragma unroll
        for (int r = 0; r < 8; ++r)
          o[(size_t)(mrow0 + r) * ldo + n] = acc[mt][nt][r];
      }
    }
  }
}

// ---------------------------------------------------------------------------
// In-place row softmax over rows of length SEQ; one wave per row.
// ---------------------------------------------------------------------------
__global__ __launch_bounds__(256) void softmax_rows(float* __restrict__ attn) {
  const int row  = blockIdx.x * 8 + (threadIdx.x >> 5);
  const int lane = threadIdx.x & 31;
  float* p = attn + (size_t)row * SEQ;

  v4f v[16];
  float mx = -__builtin_inff();
#pragma unroll
  for (int i = 0; i < 16; ++i) {
    v[i] = *(const v4f*)(p + (i * 32 + lane) * 4);
#pragma unroll
    for (int r = 0; r < 4; ++r) mx = fmaxf(mx, v[i][r]);
  }
#pragma unroll
  for (int off = 16; off > 0; off >>= 1) mx = fmaxf(mx, __shfl_xor(mx, off, 32));

  float sum = 0.f;
#pragma unroll
  for (int i = 0; i < 16; ++i)
#pragma unroll
    for (int r = 0; r < 4; ++r) {
      float e = __expf(v[i][r] - mx);
      v[i][r] = e;
      sum += e;
    }
#pragma unroll
  for (int off = 16; off > 0; off >>= 1) sum += __shfl_xor(sum, off, 32);

  const float inv = 1.0f / sum;
#pragma unroll
  for (int i = 0; i < 16; ++i) {
#pragma unroll
    for (int r = 0; r < 4; ++r) v[i][r] *= inv;
    *(v4f*)(p + (i * 32 + lane) * 4) = v[i];
  }
}

// ---------------------------------------------------------------------------
extern "C" void kernel_launch(void* const* d_in, const int* in_sizes, int n_in,
                              void* d_out, int out_size, void* d_ws,
                              size_t ws_size, hipStream_t stream) {
  const float* query = (const float*)d_in[0];
  const float* key_  = (const float*)d_in[1];
  const float* value = (const float*)d_in[2];
  const int*   mask  = (const int*)d_in[3];
  const float* wq_w  = (const float*)d_in[4];
  const float* wq_b  = (const float*)d_in[5];
  const float* wk_w  = (const float*)d_in[6];
  const float* wk_b  = (const float*)d_in[7];
  const float* wv_w  = (const float*)d_in[8];
  const float* wv_b  = (const float*)d_in[9];

  float* ctx_out  = (float*)d_out;                              // [B][S][D]
  float* attn_out = (float*)d_out + (size_t)BATCH * SEQ * DIMN; // [B][S][S]

  // workspace: Q, K bf16 [B][S][D]; V bf16 transposed [B][D][S]  (48 MB)
  bf16* Qb = (bf16*)d_ws;
  bf16* Kb = Qb + (size_t)BATCH * SEQ * DIMN;
  bf16* Vt = Kb + (size_t)BATCH * SEQ * DIMN;

  dim3 blk(256);

  // 1-3) projections: M = B*S = 8192, N = DIM, K = DIM
  dim3 gproj(DIMN / BN, (BATCH * SEQ) / BM, 1);
  gemm_wmma<true, true, EPI_BF16_OUT><<<gproj, blk, 0, stream>>>(
      query, wq_w, Qb, DIMN, DIMN, DIMN, DIMN, 0, 0, 0, wq_b, nullptr, 0, 1.f);
  gemm_wmma<true, true, EPI_BF16_OUT><<<gproj, blk, 0, stream>>>(
      key_, wk_w, Kb, DIMN, DIMN, DIMN, DIMN, 0, 0, 0, wk_b, nullptr, 0, 1.f);
  gemm_wmma<true, true, EPI_BF16_T_OUT><<<gproj, blk, 0, stream>>>(
      value, wv_w, Vt, DIMN, DIMN, DIMN, DIMN, 0, 0, 0, wv_b, nullptr, 0, 1.f);

  // 4) scores: per batch 2048 x 2048, K = 1024; scale 1/sqrt(1024), mask->-inf
  dim3 gsc(SEQ / BN, SEQ / BM, BATCH);
  gemm_wmma<false, false, EPI_SCORE_OUT><<<gsc, blk, 0, stream>>>(
      Qb, Kb, attn_out, DIMN, DIMN, SEQ, DIMN, (long)SEQ * DIMN,
      (long)SEQ * DIMN, (long)SEQ * SEQ, nullptr, mask, SEQ, 1.0f / 32.0f);

  // 5) softmax in place over attn rows
  softmax_rows<<<(BATCH * SEQ) / 8, blk, 0, stream>>>(attn_out);

  // 6) context: per batch M = 2048 (q), N = 1024 (d), K = 2048 (k)
  dim3 gctx(DIMN / BN, SEQ / BM, BATCH);
  gemm_wmma<true, false, EPI_F32_OUT><<<gctx, blk, 0, stream>>>(
      attn_out, Vt, ctx_out, SEQ, SEQ, DIMN, SEQ, (long)SEQ * SEQ,
      (long)DIMN * SEQ, (long)SEQ * DIMN, nullptr, nullptr, 0, 1.f);
}
